// Seq2SeqModel_80736795230381
// MI455X (gfx1250) — compile-verified, matrix operand order checked by reading, not run on hardware
//
#include <hip/hip_runtime.h>

// ---------------- constants ----------------
#define Vn   32001
#define Dn   512
#define Hn   1024
#define Bn   32
#define TSn  64
#define TDn  64
#define G4H  4096       // 4*H
#define ROWS 2048       // T*B rows (time-major)

#define LSTM_WGS   32
#define LSTM_THR   128                      // 4 waves: (mt 0..1) x (jloc 0..1)
#define LSTM_LDS_B (8 * 16 * Hn * 2)        // 8 B-tiles x 16 rows x K=1024 bf16 = 256KB

#define LSM_LDS_B  (32032 * 4)              // one padded logits row in LDS = 128KB

typedef __attribute__((ext_vector_type(16))) __bf16 v16bf;
typedef __attribute__((ext_vector_type(8)))  __bf16 v8bf;
typedef __attribute__((ext_vector_type(8)))  float  v8f;

// Load a 16-bf16 WMMA A/B fragment for one lane (CDNA5 §7.12.2 16-bit layout):
// lane r=lane&15 holds row/col r; k-half=(lane>>4)*8; regs = K{kh..kh+7},{kh+16..kh+23}.
__device__ __forceinline__ v16bf ld_frag(const __bf16* p) {
  const v8bf* q = reinterpret_cast<const v8bf*>(p);
  v8bf lo = q[0];
  v8bf hi = q[2];   // +16 elements
  v16bf r;
#pragma unroll
  for (int i = 0; i < 8; ++i) { r[i] = lo[i]; r[i + 8] = hi[i]; }
  return r;
}

__device__ __forceinline__ v8f wmma_bf16(v16bf a, v16bf b, v8f c) {
  return __builtin_amdgcn_wmma_f32_16x16x32_bf16(false, a, false, b,
                                                 (short)0, c, false, false);
}

__device__ __forceinline__ float sigf(float x) {
  return 1.0f / (1.0f + __expf(-x));
}

// ---------------- grid-wide barrier ----------------
__device__ __forceinline__ void gsync(unsigned* cnt, unsigned* gen) {
  __syncthreads();
  if (threadIdx.x == 0) {
    __threadfence();
    volatile unsigned* vgen = (volatile unsigned*)gen;
    unsigned g = *vgen;
    if (atomicAdd(cnt, 1u) == gridDim.x - 1) {
      atomicExch(cnt, 0u);
      __threadfence();
      atomicAdd(gen, 1u);
    } else {
      while (*vgen == g) { __builtin_amdgcn_s_sleep(1); }
    }
    __threadfence();
  }
  __syncthreads();
}

__global__ void k_init_bars(unsigned* b) { b[0] = 0u; b[1] = 0u; }

// ---------------- f32 -> bf16 convert ----------------
__global__ void k_cvt_bf16(const float* __restrict__ s, __bf16* __restrict__ d, int n) {
  int i = blockIdx.x * blockDim.x + threadIdx.x;
  if (i < n) d[i] = (__bf16)s[i];
}

// ---------------- embedding gather (batch-first tokens -> time-major bf16) ----
__global__ void k_embed(const int* __restrict__ tok, const float* __restrict__ emb,
                        __bf16* __restrict__ out, int T) {
  int i = blockIdx.x * blockDim.x + threadIdx.x;     // (t*B+b)*D + d
  int total = T * Bn * Dn;
  if (i >= total) return;
  int d  = i % Dn;
  int tb = i / Dn;
  int b  = tb % Bn;
  int t  = tb / Bn;
  int tk = tok[b * T + t];
  out[i] = (__bf16)emb[(long)tk * Dn + d];
}

// ---------------- WMMA GEMM: C[M,N] = A[M,K] * W[N,K]^T (+bias), f32 out ------
// block = 128 threads (4 waves); block tile 16 x 256; wave tile 16 x 64.
__global__ void k_gemm_xproj(const __bf16* __restrict__ A, const __bf16* __restrict__ W,
                             const float* __restrict__ bias, float* __restrict__ C,
                             int M, int N, int K) {
  const int lane = threadIdx.x & 31;
  const int wave = threadIdx.x >> 5;
  const int m0 = blockIdx.x * 16;
  const int n0 = blockIdx.y * 256 + wave * 64;
  const int r = lane & 15, half = lane >> 4;
  const __bf16* arow = A + (long)(m0 + r) * K + half * 8;
  const __bf16* wrow = W + (long)(n0 + r) * K + half * 8;
  v8f z = {};
  v8f acc[4] = { z, z, z, z };
  for (int k = 0; k < K; k += 32) {
    v16bf a = ld_frag(arow + k);
#pragma unroll
    for (int j = 0; j < 4; ++j) {
      v16bf b = ld_frag(wrow + (long)j * 16 * K + k);
      acc[j] = wmma_bf16(a, b, acc[j]);
    }
  }
  const int cl = lane & 15, rb = (lane >> 4) * 8;
#pragma unroll
  for (int j = 0; j < 4; ++j) {
    int n = n0 + j * 16 + cl;
    float bv = bias ? bias[n] : 0.0f;
#pragma unroll
    for (int rr = 0; rr < 8; ++rr) {
      C[(long)(m0 + rb + rr) * N + n] = acc[j][rr] + bv;
    }
  }
}

// ---------------- fused persistent LSTM layer --------------------------------
// grid = 32 WGs x 128 threads (4 waves). Wave (mt,jloc) owns batch rows
// mt*16..+15 and hidden cols j0..j0+15, computes ALL FOUR gates for its patch
// (4 WMMA accumulators seeded with Xp[t]), keeps cell state c in registers for
// the whole layer, and chains h through Yx slots: step t reads Yx[t], writes
// Yx[t+1]. Whh fragments for this WG (8 tiles of 16x1024 bf16 = 256KB) are
// staged once into LDS; only ONE grid barrier per timestep.
extern __shared__ __bf16 Blds[];   // LSTM_LDS_B bytes

__global__ void k_lstm_fused(const float* __restrict__ Xp,     // [T*B, 4H]
                             const __bf16* __restrict__ Whh,   // [4H, H]
                             const __bf16* __restrict__ h_init,// [B,H] or null
                             const float*  __restrict__ c_init,// [B,H] or null
                             __bf16* __restrict__ Yx,          // [(T+1)*B, H]
                             float*  __restrict__ cT,          // [B,H] or null
                             int T, unsigned* bar_cnt, unsigned* bar_gen) {
  const int lane = threadIdx.x & 31;
  const int wave = threadIdx.x >> 5;       // 0..3
  const int mt   = wave >> 1;              // batch tile 0..1
  const int jloc = wave & 1;               // local hidden tile 0..1
  const int m0 = mt * 16;
  const int j0 = (blockIdx.x * 2 + jloc) * 16;
  const int r = lane & 15, half = lane >> 4;
  const int cl = lane & 15, rb = (lane >> 4) * 8;
  const int gtid = blockIdx.x * LSTM_THR + threadIdx.x;

  // ---- stage Whh tiles (rows n = g*H + j0base + r, full K) into LDS ----
  // LDS layout: [(tileIdx*16 + row)*Hn + k], tileIdx = g*2 + jl.
  for (int ci = threadIdx.x; ci < (8 * 16 * Hn) / 8; ci += LSTM_THR) {
    int elem = ci * 8;
    int row  = elem >> 10;               // 0..127
    int k    = elem & (Hn - 1);
    int tIdx = row >> 4;
    int rr   = row & 15;
    int g    = tIdx >> 1;
    int jl   = tIdx & 1;
    int n    = g * Hn + (blockIdx.x * 2 + jl) * 16 + rr;
    *(v8bf*)&Blds[row * Hn + k] =
        *(const v8bf*)(Whh + (long)n * Hn + k);
  }

  // ---- initial h into slot 0, initial c into registers ----
  for (int i = gtid; i < Bn * Hn; i += LSTM_WGS * LSTM_THR)
    Yx[i] = h_init ? h_init[i] : (__bf16)0.0f;
  v8f c_frag = {};
  if (c_init) {
#pragma unroll
    for (int rr = 0; rr < 8; ++rr)
      c_frag[rr] = c_init[((m0 + rb + rr) << 10) + j0 + cl];
  }
  gsync(bar_cnt, bar_gen);   // also covers the LDS staging via its syncthreads

  const unsigned bbase = (unsigned)jloc * 16 * Hn + (unsigned)r * Hn + half * 8;

  for (int t = 0; t < T; ++t) {
    // seed accumulators with Xp[t] (bias already folded in by xproj GEMM)
    const float* xpb = Xp + ((long)t * Bn + m0 + rb) * G4H + j0 + cl;
    v8f acc[4];
#pragma unroll
    for (int g = 0; g < 4; ++g)
#pragma unroll
      for (int rr = 0; rr < 8; ++rr)
        acc[g][rr] = xpb[(long)rr * G4H + g * Hn];

    // gate GEMM: acc[g] += h(t) @ Whh[g-block]^T   (A from global, B from LDS)
    const __bf16* ar = Yx + (long)t * (Bn * Hn) + ((long)(m0 + r) << 10) + half * 8;
    for (int k = 0; k < Hn; k += 32) {
      v16bf a = ld_frag(ar + k);
#pragma unroll
      for (int g = 0; g < 4; ++g) {
        v16bf b = ld_frag(&Blds[bbase + (unsigned)g * (32 * Hn) + k]);
        acc[g] = wmma_bf16(a, b, acc[g]);
      }
    }

    // fused cell update (c in registers), h -> slot t+1
    __bf16* yo = Yx + (long)(t + 1) * (Bn * Hn);
#pragma unroll
    for (int rr = 0; rr < 8; ++rr) {
      float ii = acc[0][rr];
      float ff = acc[1][rr];
      float gg = acc[2][rr];
      float oo = acc[3][rr];
      float c = sigf(ff) * c_frag[rr] + sigf(ii) * tanhf(gg);
      float h = sigf(oo) * tanhf(c);
      c_frag[rr] = c;
      yo[((long)(m0 + rb + rr) << 10) + j0 + cl] = (__bf16)h;
    }
    if (t + 1 < T)
      __builtin_prefetch(Xp + ((long)(t + 1) * Bn + m0 + rb) * G4H + j0 + cl, 0, 1);
    gsync(bar_cnt, bar_gen);   // h(t+1) visible device-wide before next step
  }

  if (cT) {
#pragma unroll
    for (int rr = 0; rr < 8; ++rr)
      cT[((m0 + rb + rr) << 10) + j0 + cl] = c_frag[rr];
  }
}

// ---------------- logits GEMM: out[b,t,:] = D1[t*B+b,:] @ Wpred^T + b_pred ----
__global__ void k_gemm_logits(const __bf16* __restrict__ A,   // [2048, H]
                              const __bf16* __restrict__ Wp,  // [V, H]
                              const float* __restrict__ bias, // [V]
                              float* __restrict__ out) {
  const int lane = threadIdx.x & 31;
  const int wave = threadIdx.x >> 5;
  const int m0 = blockIdx.x * 16;
  const int n0 = blockIdx.y * 256 + wave * 64;
  const int r = lane & 15, half = lane >> 4;
  const __bf16* arow = A + (long)(m0 + r) * Hn + half * 8;
  const __bf16* wbase[4];
#pragma unroll
  for (int j = 0; j < 4; ++j) {
    int nr = n0 + j * 16 + r;
    if (nr >= Vn) nr = Vn - 1;                 // clamp tail reads
    wbase[j] = Wp + (long)nr * Hn + half * 8;
  }
  v8f z = {};
  v8f acc[4] = { z, z, z, z };
  for (int k = 0; k < Hn; k += 32) {
    v16bf a = ld_frag(arow + k);
#pragma unroll
    for (int j = 0; j < 4; ++j) {
      v16bf b = ld_frag(wbase[j] + k);
      acc[j] = wmma_bf16(a, b, acc[j]);
    }
  }
  const int cl = lane & 15, rb = (lane >> 4) * 8;
#pragma unroll
  for (int j = 0; j < 4; ++j) {
    int n = n0 + j * 16 + cl;
    if (n >= Vn) continue;
    float bv = bias[n];
#pragma unroll
    for (int rr = 0; rr < 8; ++rr) {
      int m = m0 + rb + rr;                    // row = t*B + b
      int t = m >> 5;
      int b = m & 31;
      out[(long)b * TDn * Vn + (long)t * Vn + n] = acc[j][rr] + bv;
    }
  }
}

// ---------------- in-place row log-softmax (row staged in 128KB LDS) ---------
// One 256-thread WG per (b,t) row. The 32001-float row is read from HBM ONCE
// into LDS (fits easily in the 320KB WGP LDS), both reductions run out of LDS,
// and the normalized row is written back once: 2 passes of HBM traffic not 4.
extern __shared__ float Lrow[];   // LSM_LDS_B bytes

__device__ __forceinline__ float wave_max(float v) {
#pragma unroll
  for (int o = 16; o; o >>= 1) v = fmaxf(v, __shfl_xor(v, o, 32));
  return v;
}
__device__ __forceinline__ float wave_sum(float v) {
#pragma unroll
  for (int o = 16; o; o >>= 1) v += __shfl_xor(v, o, 32);
  return v;
}

__global__ void k_logsoftmax(float* __restrict__ out) {
  __shared__ float sm[8];
  float* p = out + (long)blockIdx.x * Vn;
  const int w = threadIdx.x >> 5;

  // stage row into LDS, folding the max reduction into the same pass
  float m = -1e30f;
  for (int i = threadIdx.x; i < Vn; i += blockDim.x) {
    float v = p[i];
    Lrow[i] = v;
    m = fmaxf(m, v);
  }
  m = wave_max(m);
  if ((threadIdx.x & 31) == 0) sm[w] = m;
  __syncthreads();
  float mAll = -1e30f;
#pragma unroll
  for (int i = 0; i < 8; ++i) mAll = fmaxf(mAll, sm[i]);
  __syncthreads();

  float s = 0.0f;
  for (int i = threadIdx.x; i < Vn; i += blockDim.x) s += __expf(Lrow[i] - mAll);
  s = wave_sum(s);
  if ((threadIdx.x & 31) == 0) sm[w] = s;
  __syncthreads();
  float sAll = 0.0f;
#pragma unroll
  for (int i = 0; i < 8; ++i) sAll += sm[i];
  float lse = mAll + __logf(sAll);

  for (int i = threadIdx.x; i < Vn; i += blockDim.x) p[i] = Lrow[i] - lse;
}

// ---------------- host orchestration ----------------------------------------
extern "C" void kernel_launch(void* const* d_in, const int* in_sizes, int n_in,
                              void* d_out, int out_size, void* d_ws, size_t ws_size,
                              hipStream_t stream) {
  const int*   src      = (const int*)  d_in[0];
  const int*   dst      = (const int*)  d_in[1];
  const float* emb_in   = (const float*)d_in[2];
  const float* emb_out  = (const float*)d_in[3];
  const float* eWih0    = (const float*)d_in[4];
  const float* eWhh0    = (const float*)d_in[5];
  const float* eb0      = (const float*)d_in[6];
  const float* eWih1    = (const float*)d_in[7];
  const float* eWhh1    = (const float*)d_in[8];
  const float* eb1      = (const float*)d_in[9];
  const float* dWih0    = (const float*)d_in[10];
  const float* dWhh0    = (const float*)d_in[11];
  const float* dWih1    = (const float*)d_in[12];
  const float* dWhh1    = (const float*)d_in[13];
  const float* W_pred   = (const float*)d_in[14];
  const float* b_pred   = (const float*)d_in[15];
  float* out = (float*)d_out;

  char* wp = (char*)d_ws;
  auto alloc = [&](size_t bytes) -> char* {
    char* p = wp;
    wp += (bytes + 255) & ~(size_t)255;
    return p;
  };
  unsigned* bars   = (unsigned*)alloc(256);
  __bf16* Xsrc     = (__bf16*)alloc((size_t)ROWS * Dn * 2);
  __bf16* E        = (__bf16*)alloc((size_t)ROWS * Dn * 2);
  __bf16* Wih_bf   = (__bf16*)alloc((size_t)G4H * Hn * 2);
  __bf16* Whh_bf   = (__bf16*)alloc((size_t)G4H * Hn * 2);
  __bf16* Wpred_bf = (__bf16*)alloc((size_t)Vn * Hn * 2);
  float*  Xp       = (float*) alloc((size_t)ROWS * G4H * 4);
  float*  c0f      = (float*) alloc((size_t)Bn * Hn * 4);
  float*  c1f      = (float*) alloc((size_t)Bn * Hn * 4);
  __bf16* Yx_a     = (__bf16*)alloc((size_t)(TSn + 1) * Bn * Hn * 2);  // enc0 / dec0
  __bf16* Yx_b     = (__bf16*)alloc((size_t)(TSn + 1) * Bn * Hn * 2);  // enc1 / dec1

  auto cvt = [&](const float* s, __bf16* d, int n) {
    k_cvt_bf16<<<(n + 255) / 256, 256, 0, stream>>>(s, d, n);
  };

  (void)hipFuncSetAttribute(reinterpret_cast<const void*>(k_lstm_fused),
                            hipFuncAttributeMaxDynamicSharedMemorySize, LSTM_LDS_B);
  (void)hipFuncSetAttribute(reinterpret_cast<const void*>(k_logsoftmax),
                            hipFuncAttributeMaxDynamicSharedMemorySize, LSM_LDS_B);

  k_init_bars<<<1, 1, 0, stream>>>(bars);
  cvt(W_pred, Wpred_bf, Vn * Hn);
  k_embed<<<(TSn * Bn * Dn + 255) / 256, 256, 0, stream>>>(src, emb_in,  Xsrc, TSn);
  k_embed<<<(TDn * Bn * Dn + 255) / 256, 256, 0, stream>>>(dst, emb_out, E,    TDn);

  dim3 gx(ROWS / 16, G4H / 256);
  unsigned* bc = bars;
  unsigned* bg = bars + 1;
  const int SLOT = Bn * Hn;

  // ---- encoder layer 0 ----
  cvt(eWih0, Wih_bf, G4H * Dn);
  cvt(eWhh0, Whh_bf, G4H * Hn);
  k_gemm_xproj<<<gx, 128, 0, stream>>>(Xsrc, Wih_bf, eb0, Xp, ROWS, G4H, Dn);
  k_lstm_fused<<<LSTM_WGS, LSTM_THR, LSTM_LDS_B, stream>>>(
      Xp, Whh_bf, nullptr, nullptr, Yx_a, c0f, TSn, bc, bg);
  // ---- encoder layer 1 (only final state consumed downstream) ----
  cvt(eWih1, Wih_bf, G4H * Hn);
  cvt(eWhh1, Whh_bf, G4H * Hn);
  k_gemm_xproj<<<gx, 128, 0, stream>>>(Yx_a + SLOT, Wih_bf, eb1, Xp, ROWS, G4H, Hn);
  k_lstm_fused<<<LSTM_WGS, LSTM_THR, LSTM_LDS_B, stream>>>(
      Xp, Whh_bf, nullptr, nullptr, Yx_b, c1f, TSn, bc, bg);
  // ---- decoder layer 0 (bridged from enc layer-0 final state; reuses Yx_a) --
  cvt(dWih0, Wih_bf, G4H * Dn);
  cvt(dWhh0, Whh_bf, G4H * Hn);
  k_gemm_xproj<<<gx, 128, 0, stream>>>(E, Wih_bf, nullptr, Xp, ROWS, G4H, Dn);
  k_lstm_fused<<<LSTM_WGS, LSTM_THR, LSTM_LDS_B, stream>>>(
      Xp, Whh_bf, Yx_a + (long)TSn * SLOT, c0f, Yx_a, nullptr, TDn, bc, bg);
  // ---- decoder layer 1 (bridged from enc layer-1 final state; reuses Yx_b) --
  cvt(dWih1, Wih_bf, G4H * Hn);
  cvt(dWhh1, Whh_bf, G4H * Hn);
  k_gemm_xproj<<<gx, 128, 0, stream>>>(Yx_a + SLOT, Wih_bf, nullptr, Xp, ROWS, G4H, Hn);
  k_lstm_fused<<<LSTM_WGS, LSTM_THR, LSTM_LDS_B, stream>>>(
      Xp, Whh_bf, Yx_b + (long)TSn * SLOT, c1f, Yx_b, nullptr, TDn, bc, bg);

  // ---- logits + in-place log-softmax ----
  dim3 gl(ROWS / 16, (Vn + 255) / 256);
  k_gemm_logits<<<gl, 128, 0, stream>>>(Yx_b + SLOT, Wpred_bf, b_pred, out);
  k_logsoftmax<<<ROWS, 256, LSM_LDS_B, stream>>>(out);
}
